// E2EStackedBiRNNMpPoolAttentionWithoutWordEmb_82437602279864
// MI455X (gfx1250) — compile-verified, matrix operand order checked by reading, not run on hardware
//
#include <hip/hip_runtime.h>
#include <hip/hip_bf16.h>

// ---------------------------------------------------------------------------
// Problem constants
// ---------------------------------------------------------------------------
#define Bq   48
#define Tq   128
#define DWq  320
#define Uq   256
#define Pq   256
#define DINq 322
#define GNq  768          // 3*U
#define BTq  (Bq*Tq)      // 6144

typedef __attribute__((ext_vector_type(16))) __bf16 v16bf;
typedef __attribute__((ext_vector_type(8)))  float  v8f;

// f32 -> bf16, round-to-nearest-even via bit manipulation
__device__ __forceinline__ __bf16 f2bf(float f) {
    union { float f; unsigned u; } v; v.f = f;
    unsigned r = (v.u + 0x7FFFu + ((v.u >> 16) & 1u)) >> 16;
    union { unsigned short s; __bf16 b; } o; o.s = (unsigned short)r;
    return o.b;
}

// CDNA5 16-bit A/B fragment k-index for 16x16x32 (CDNA5 ISA §7.12.2):
//   VGPR v packs K=(k0,k0+1), k0 = (v&3)*2 + (v>>2)*16 + half*8
__device__ __forceinline__ int frag_k(int lane, int e) {
    int v = e >> 1, lo = e & 1, half = lane >> 4;
    return ((v & 3) * 2) + ((v >> 2) * 16) + half * 8 + lo;
}

// ---------------------------------------------------------------------------
// Generic fragment packer: strided f32 matrix -> bf16 WMMA fragments.
//   element(b, r, k) = src[b*sbs + r*rs + k*ks], zero for k >= K
//   out slot (v16bf): [b*RT*KT*32 + ((rt*KT + kt)*32 + lane)]
// ---------------------------------------------------------------------------
__global__ __launch_bounds__(256)
void pack_frag(const float* __restrict__ src, __bf16* __restrict__ out,
               int RT, int K, int KT, long rs, long ks, long sbs)
{
    int idx = blockIdx.x * blockDim.x + threadIdx.x;
    int total = RT * KT * 32;
    if (idx >= total) return;
    int lane = idx & 31;
    int kt   = (idx >> 5) % KT;
    int rt   = (idx >> 5) / KT;
    long b   = blockIdx.z;
    const float* S = src + b * sbs;
    int r = rt * 16 + (lane & 15);
    v16bf tmp;
#pragma unroll
    for (int e = 0; e < 16; ++e) {
        int k  = kt * 32 + frag_k(lane, e);
        int kc = (k < K) ? k : 0;                  // clamp -> unconditional load
        float v = S[(long)r * rs + (long)kc * ks];
        tmp[e] = f2bf((k < K) ? v : 0.0f);         // select, no branch
    }
    *((v16bf*)out + (long)b * total + idx) = tmp;
}

// ---------------------------------------------------------------------------
// Packed WMMA GEMM. One wave computes a 16x64 strip (4 n-tiles), A fragment
// reused 4x. Inner loop is software-pipelined: next k-step's 5 fragments are
// prefetched into distinct registers before the current wmmas issue, so the
// compiler can overlap load latency with wmma via partial loadcnt waits.
// ---------------------------------------------------------------------------
__global__ __launch_bounds__(256)
void wmma_gemm_packed(const __bf16* __restrict__ PA, const __bf16* __restrict__ PW,
                      const float* __restrict__ bias, float* __restrict__ C,
                      int tiles_m, int tiles_n, int KT,
                      long ldc, long c_bs, int relu)
{
    int wave = (int)((blockIdx.x * blockDim.x + threadIdx.x) >> 5);
    int lane = (int)(threadIdx.x & 31);
    int tn4  = tiles_n >> 2;
    int jobs = tiles_m * tn4;
    if (wave >= jobs) return;
    int tm = wave / tn4, tg = wave % tn4;
    long b = blockIdx.z;

    const v16bf* af = (const v16bf*)PA + ((long)(b * tiles_m + tm) * KT) * 32 + lane;
    const v16bf* wf = (const v16bf*)PW + ((long)(b * tiles_n + tg * 4) * KT) * 32 + lane;
    const long wstep = (long)KT * 32;

    v8f acc[4] = {{}, {}, {}, {}};
    v16bf a_c  = af[0];
    v16bf w_c0 = wf[0];
    v16bf w_c1 = wf[wstep];
    v16bf w_c2 = wf[2 * wstep];
    v16bf w_c3 = wf[3 * wstep];
    for (int kt = 0; kt < KT; ++kt) {
        long o = (long)((kt + 1 < KT) ? kt + 1 : kt) * 32;   // clamped prefetch
        v16bf a_n  = af[o];
        v16bf w_n0 = wf[o];
        v16bf w_n1 = wf[o + wstep];
        v16bf w_n2 = wf[o + 2 * wstep];
        v16bf w_n3 = wf[o + 3 * wstep];
        acc[0] = __builtin_amdgcn_wmma_f32_16x16x32_bf16(false, a_c, false, w_c0, (short)0, acc[0], false, false);
        acc[1] = __builtin_amdgcn_wmma_f32_16x16x32_bf16(false, a_c, false, w_c1, (short)0, acc[1], false, false);
        acc[2] = __builtin_amdgcn_wmma_f32_16x16x32_bf16(false, a_c, false, w_c2, (short)0, acc[2], false, false);
        acc[3] = __builtin_amdgcn_wmma_f32_16x16x32_bf16(false, a_c, false, w_c3, (short)0, acc[3], false, false);
        a_c = a_n; w_c0 = w_n0; w_c1 = w_n1; w_c2 = w_n2; w_c3 = w_n3;
    }

    int half = lane >> 4, nl = lane & 15;
    float* Cb = C + b * c_bs;
#pragma unroll
    for (int j = 0; j < 4; ++j) {
        int n = (tg * 4 + j) * 16 + nl;
        float bb = bias ? bias[n] : 0.0f;
#pragma unroll
        for (int v = 0; v < 8; ++v) {
            int row = tm * 16 + v + half * 8;
            float o = acc[j][v] + bb;
            if (relu) o = fmaxf(o, 0.0f);
            Cb[(long)row * ldc + n] = o;
        }
    }
}

// ---------------------------------------------------------------------------
// Persistent GRU scan. 3 workgroups; WG wg owns batch rows [wg*16, wg*16+16).
// 512 threads = 16 waves; wave w computes n-tiles {3w,3w+1,3w+2} of gh.
// The 24 wmma / step are a single fully-unrolled pipeline: A fragments are
// register-resident; W fragments stream from L2 double-buffered one ahead.
// ---------------------------------------------------------------------------
__global__ __launch_bounds__(512)
void gru_scan(const float* __restrict__ xp,      // (B,T,768), includes bih
              const __bf16* __restrict__ wfrag,  // packed Whh fragments
              const float* __restrict__ bhh,     // (768)
              const float* __restrict__ resid,   // (B,T,U) or nullptr
              float* __restrict__ out)           // (B,T,U)
{
    __shared__ float hbuf[16][Uq];
    __shared__ float gh[16][GNq];
    const int KT = Uq / 32;    // 8
    int wg   = blockIdx.x;
    int tid  = (int)threadIdx.x;
    int wave = tid >> 5, lane = tid & 31;
    int mrow = lane & 15, half = lane >> 4;

    for (int i = tid; i < 16 * Uq; i += (int)blockDim.x)
        (&hbuf[0][0])[i] = 0.0f;
    __syncthreads();

    const v16bf* wbase = (const v16bf*)wfrag + lane;

    for (int t = 0; t < Tq; ++t) {
        // build all 8 A fragments (h -> bf16) once per step
        v16bf afrag[KT];
#pragma unroll
        for (int kt = 0; kt < KT; ++kt)
#pragma unroll
            for (int e = 0; e < 16; ++e)
                afrag[kt][e] = f2bf(hbuf[mrow][kt * 32 + frag_k(lane, e)]);

        // 24-step pipelined wmma chain (3 n-tiles x 8 k-steps)
        v8f acc[3] = {{}, {}, {}};
        v16bf w_c = wbase[(long)(wave * 3) * KT * 32];
#pragma unroll
        for (int it = 0; it < 3 * KT; ++it) {
            int itn = (it + 1 < 3 * KT) ? it + 1 : it;       // clamped prefetch
            int jn = itn / KT, ktn = itn % KT;
            v16bf w_n = wbase[((long)(wave * 3 + jn) * KT + ktn) * 32];
            int j = it / KT, kt = it % KT;
            acc[j] = __builtin_amdgcn_wmma_f32_16x16x32_bf16(false, afrag[kt], false, w_c,
                                                             (short)0, acc[j], false, false);
            w_c = w_n;
        }
#pragma unroll
        for (int j = 0; j < 3; ++j) {
            int n = (wave * 3 + j) * 16 + (lane & 15);
#pragma unroll
            for (int v = 0; v < 8; ++v)
                gh[v + half * 8][n] = acc[j][v];
        }
        __syncthreads();

        // gate phase: 16*256 elements / 512 threads
        for (int i = tid; i < 16 * Uq; i += (int)blockDim.x) {
            int m = i >> 8, u = i & 255;
            int b = wg * 16 + m;
            long xbase = ((long)b * Tq + t) * GNq;
            float gr = gh[m][u]          + bhh[u]          + xp[xbase + u];
            float gz = gh[m][Uq + u]     + bhh[Uq + u]     + xp[xbase + Uq + u];
            float gn = gh[m][2 * Uq + u] + bhh[2 * Uq + u];
            float xn = xp[xbase + 2 * Uq + u];
            float r  = 1.0f / (1.0f + __expf(-gr));
            float z  = 1.0f / (1.0f + __expf(-gz));
            float nv = tanhf(xn + r * gn);
            float h  = hbuf[m][u];
            float hn = (1.0f - z) * nv + z * h;
            hbuf[m][u] = hn;
            long oidx = ((long)b * Tq + t) * Uq + u;
            out[oidx] = resid ? (resid[oidx] + hn) : hn;
        }
        __syncthreads();
    }
}

// ---------------------------------------------------------------------------
// Small elementwise / reduction kernels
// ---------------------------------------------------------------------------
__global__ void concat_in(const float* __restrict__ e, const float* __restrict__ p,
                          const float* __restrict__ tg, float* __restrict__ x0, int total)
{
    int i = blockIdx.x * blockDim.x + threadIdx.x;
    if (i >= total) return;
    int d = i % DINq, bt = i / DINq;
    float v;
    if (d < DWq)        v = e[(long)bt * DWq + d];
    else if (d == DWq)  v = p[bt];
    else                v = tg[bt];
    x0[i] = v;
}

__global__ void reverse_time(const float* __restrict__ in, float* __restrict__ o, int total)
{
    int i = blockIdx.x * blockDim.x + threadIdx.x;
    if (i >= total) return;
    int c = i % Uq, t = (i / Uq) % Tq, b = i / (Uq * Tq);
    o[((long)b * Tq + t) * Uq + c] = in[((long)b * Tq + (Tq - 1 - t)) * Uq + c];
}

__global__ void hjmax_k(const float* __restrict__ hj, float* __restrict__ hm)
{
    int i = blockIdx.x * blockDim.x + threadIdx.x;
    if (i >= Tq * Pq) return;
    int p = i & 255, t = i >> 8;
    float m = -INFINITY;
    for (int b = 0; b < Bq; ++b)
        m = fmaxf(m, hj[((long)(b * Tq + t)) * Pq + p]);
    hm[i] = m;
}

// outputs = relu(hi + hjmax + lin1_b) -> combined[:, :256] (ldc=512)
__global__ void build_outputs(const float* __restrict__ hi, const float* __restrict__ hm,
                              const float* __restrict__ b1, float* __restrict__ comb)
{
    int i = blockIdx.x * blockDim.x + threadIdx.x;
    if (i >= BTq * Pq) return;
    int p = i & 255, bt = i >> 8, t = bt & (Tq - 1);
    float v = hi[i] + hm[t * Pq + p] + b1[p];
    comb[(long)bt * 512 + p] = fmaxf(v, 0.0f);
}

__global__ void softmax_rows(float* __restrict__ s, int rows, int n)
{
    int r = blockIdx.x * blockDim.x + threadIdx.x;
    if (r >= rows) return;
    float* row = s + (long)r * n;
    float mx = -INFINITY;
    for (int i = 0; i < n; ++i) mx = fmaxf(mx, row[i]);
    float sum = 0.0f;
    for (int i = 0; i < n; ++i) { float e = __expf(row[i] - mx); row[i] = e; sum += e; }
    float inv = 1.0f / sum;
    for (int i = 0; i < n; ++i) row[i] *= inv;
}

// logits (N=4) + log_softmax, one thread per (b,t) row
__global__ void head_k(const float* __restrict__ hidden, const float* __restrict__ oW,
                       const float* __restrict__ ob, float* __restrict__ outp)
{
    int r = blockIdx.x * blockDim.x + threadIdx.x;
    if (r >= BTq) return;
    float lg[4];
    const float* h = hidden + (long)r * Pq;
#pragma unroll
    for (int o = 0; o < 4; ++o) {
        float acc = ob[o];
        const float* w = oW + o * Pq;
        for (int k = 0; k < Pq; ++k) acc += h[k] * w[k];
        lg[o] = acc;
    }
    float mx = fmaxf(fmaxf(lg[0], lg[1]), fmaxf(lg[2], lg[3]));
    float s = 0.0f;
#pragma unroll
    for (int o = 0; o < 4; ++o) s += __expf(lg[o] - mx);
    float lse = mx + __logf(s);
#pragma unroll
    for (int o = 0; o < 4; ++o) outp[(long)r * 4 + o] = lg[o] - lse;
}

// ---------------------------------------------------------------------------
// Orchestration
// ---------------------------------------------------------------------------
extern "C" void kernel_launch(void* const* d_in, const int* in_sizes, int n_in,
                              void* d_out, int out_size, void* d_ws, size_t ws_size,
                              hipStream_t stream)
{
    const float* embeds   = (const float*)d_in[0];
    const float* is_pred  = (const float*)d_in[1];
    const float* is_tgt   = (const float*)d_in[2];
    const float* gi_Wih   = (const float*)d_in[3];
    const float* gi_Whh   = (const float*)d_in[4];
    const float* gi_bih   = (const float*)d_in[5];
    const float* gi_bhh   = (const float*)d_in[6];
    const float* gs_Wih   = (const float*)d_in[7];
    const float* gs_Whh   = (const float*)d_in[8];
    const float* gs_bih   = (const float*)d_in[9];
    const float* gs_bhh   = (const float*)d_in[10];
    const float* lin1_W   = (const float*)d_in[11];
    const float* lin1_b   = (const float*)d_in[12];
    const float* lin2_W   = (const float*)d_in[13];
    const float* lin2_b   = (const float*)d_in[14];
    const float* attn_W   = (const float*)d_in[15];
    const float* attn_b   = (const float*)d_in[16];
    const float* out_W    = (const float*)d_in[17];
    const float* out_b    = (const float*)d_in[18];
    (void)in_sizes; (void)n_in; (void)out_size; (void)ws_size;

    // ---- workspace layout (all offsets 32B aligned) ----
    char* base = (char*)d_ws;
    __bf16* packWhh = (__bf16*)base;                      // 4 * 768*256 bf16
    size_t off = (size_t)4 * GNq * Uq * sizeof(__bf16);
    __bf16* PA  = (__bf16*)(base + off); off += (size_t)BTq * 16 * 32 * sizeof(__bf16); // max KT=16
    __bf16* PW  = (__bf16*)(base + off); off += (size_t)48 * 256 * 8 * 32 * sizeof(__bf16);
    float* x0   = (float*)(base + off); off += (size_t)BTq * DINq * 4;
    float* xp   = (float*)(base + off); off += (size_t)BTq * GNq  * 4;
    float* buf0 = (float*)(base + off); off += (size_t)BTq * Uq   * 4;
    float* buf1 = (float*)(base + off); off += (size_t)BTq * Uq   * 4;
    float* cur  = (float*)(base + off); off += (size_t)BTq * Uq   * 4;
    float* hjm  = (float*)(base + off); off += (size_t)Tq  * Pq   * 4;
    float* comb = (float*)(base + off); off += (size_t)BTq * 512  * 4;

    auto pack = [&](const float* src, __bf16* dst, int R, int K,
                    long rs, long ks, long sbs, int batch) {
        int KT = (K + 31) / 32;
        int tot = (R / 16) * KT * 32;
        dim3 grid((tot + 255) / 256, 1, batch);
        pack_frag<<<grid, 256, 0, stream>>>(src, dst, R / 16, K, KT, rs, ks, sbs);
    };
    auto gemm = [&](const __bf16* pa, const __bf16* pw, const float* bias, float* C,
                    int M, int N, int K, long ldc, long c_bs, int batch, int relu) {
        int KT = (K + 31) / 32;
        int jobs = (M / 16) * (N / 64);
        dim3 grid((jobs + 7) / 8, 1, batch);
        wmma_gemm_packed<<<grid, 256, 0, stream>>>(pa, pw, bias, C,
                                                   M / 16, N / 16, KT, ldc, c_bs, relu);
    };

    // 0) concat inputs -> x0 (B,T,322)
    concat_in<<<(BTq * DINq + 255) / 256, 256, 0, stream>>>(embeds, is_pred, is_tgt,
                                                            x0, BTq * DINq);
    // 1) pack the 4 Whh matrices once (scan-resident weights)
    pack(gi_Whh, packWhh, GNq, Uq, Uq, 1, 0, 1);
    for (int l = 0; l < 3; ++l)
        pack(gs_Whh + (long)l * GNq * Uq, packWhh + (long)(l + 1) * GNq * Uq,
             GNq, Uq, Uq, 1, 0, 1);

    // 2) layer 0: xp = x0 @ Wih^T + bih ; scan
    pack(x0, PA, BTq, DINq, DINq, 1, 0, 1);
    pack(gi_Wih, PW, GNq, DINq, DINq, 1, 0, 1);
    gemm(PA, PW, gi_bih, xp, BTq, GNq, DINq, GNq, 0, 1, 0);
    gru_scan<<<3, 512, 0, stream>>>(xp, packWhh, gi_bhh, nullptr, buf0);

    // 3) layers 1..3: flip, project, scan with residual
    float* prev = buf0; float* nxt = buf1;
    for (int l = 0; l < 3; ++l) {
        reverse_time<<<(BTq * Uq + 255) / 256, 256, 0, stream>>>(prev, cur, BTq * Uq);
        pack(cur, PA, BTq, Uq, Uq, 1, 0, 1);
        pack(gs_Wih + (long)l * GNq * Uq, PW, GNq, Uq, Uq, 1, 0, 1);
        gemm(PA, PW, gs_bih + (long)l * GNq, xp, BTq, GNq, Uq, GNq, 0, 1, 0);
        gru_scan<<<3, 512, 0, stream>>>(xp, packWhh + (long)(l + 1) * GNq * Uq,
                                        gs_bhh + (long)l * GNq, cur, nxt);
        float* tswap = prev; prev = nxt; nxt = tswap;
    }
    // 4) g = reverse(out3) -> cur
    reverse_time<<<(BTq * Uq + 255) / 256, 256, 0, stream>>>(prev, cur, BTq * Uq);
    float* g  = cur;
    float* hi = xp;
    float* hj = xp + (size_t)BTq * Pq;

    // 5) hi = g@Wi^T, hj = g@Wj^T (Wi/Wj = column halves of lin1_W (P,2U))
    pack(g, PA, BTq, Uq, Uq, 1, 0, 1);
    pack(lin1_W, PW, Pq, Uq, 2 * Uq, 1, 0, 1);
    gemm(PA, PW, nullptr, hi, BTq, Pq, Uq, Pq, 0, 1, 0);
    pack(lin1_W + Uq, PW, Pq, Uq, 2 * Uq, 1, 0, 1);
    gemm(PA, PW, nullptr, hj, BTq, Pq, Uq, Pq, 0, 1, 0);

    // 6) cross-batch max (relu-monotone simplification), outputs -> comb[:, :256]
    hjmax_k<<<(Tq * Pq + 255) / 256, 256, 0, stream>>>(hj, hjm);
    build_outputs<<<(BTq * Pq + 255) / 256, 256, 0, stream>>>(hi, hjm, lin1_b, comb);

    // 7) proj = outputs @ attn_W^T + attn_b  (pack of comb[:, :256] shared with scores)
    float* proj = buf0;
    pack(comb, PA, BTq, Pq, 512, 1, 0, 1);
    pack(attn_W, PW, Pq, Pq, Pq, 1, 0, 1);
    gemm(PA, PW, attn_b, proj, BTq, Pq, Pq, Pq, 0, 1, 0);

    // 8) scores[b] = outputs[b] @ proj[b]^T (batched; reuses PA from step 7:
    //    flat tile index (b*8+tm) matches batched (b,tm) exactly)
    float* scores = buf1;
    pack(proj, PW, Tq, Pq, Pq, 1, (long)Tq * Pq, Bq);
    gemm(PA, PW, nullptr, scores, Tq, Tq, Pq, Tq, (long)Tq * Tq, Bq, 0);

    // 9) softmax over s
    softmax_rows<<<(BTq + 255) / 256, 256, 0, stream>>>(scores, BTq, Tq);

    // 10) attend[b] = alpha[b] @ outputs[b] -> comb[:, 256:512]
    pack(scores, PA, Tq, Tq, Tq, 1, (long)Tq * Tq, Bq);
    pack(comb, PW, Pq, Tq, 1, 512, (long)Tq * 512, Bq);   // W(n=p,k=s)=comb[b,s,p]
    gemm(PA, PW, nullptr, comb + Pq, Tq, Pq, Tq, 512, (long)Tq * 512, Bq, 0);

    // 11) hidden = relu(combined @ lin2_W^T + lin2_b)
    float* hidden = x0;
    pack(comb, PA, BTq, 2 * Pq, 512, 1, 0, 1);
    pack(lin2_W, PW, Pq, 2 * Pq, 2 * Pq, 1, 0, 1);
    gemm(PA, PW, lin2_b, hidden, BTq, Pq, 2 * Pq, Pq, 0, 1, 1);

    // 12) logits + log_softmax
    head_k<<<(BTq + 255) / 256, 256, 0, stream>>>(hidden, out_W, out_b, (float*)d_out);
}
